// MultiheadAttention_9070970929223
// MI455X (gfx1250) — matrix operand order repaired and checked by reading, not compiled, not run on hardware
//
#include <hip/hip_runtime.h>

#define D_MODELC 512
#define HEADSC 8
#define DKC 64
#define SEQC 4096
#define BATCHC 2

typedef _Float16 h8  __attribute__((ext_vector_type(8)));
typedef _Float16 h16 __attribute__((ext_vector_type(16)));
typedef float    f8v __attribute__((ext_vector_type(8)));

__device__ __forceinline__ f8v f8zero() {
  f8v v;
#pragma unroll
  for (int i = 0; i < 8; ++i) v[i] = 0.0f;
  return v;
}

__device__ __forceinline__ f8v wmma16(h16 a, h16 b, f8v c) {
  // D = A(16x32 f16) * B(32x16 f16) + C(16x16 f32)
  return __builtin_amdgcn_wmma_f32_16x16x32_f16(false, a, false, b, (short)0, c,
                                                false, false);
}

__device__ __forceinline__ h16 cat16(h8 a, h8 b) {
  h16 r;
#pragma unroll
  for (int i = 0; i < 8; ++i) { r[i] = a[i]; r[8 + i] = b[i]; }
  return r;
}

// A-tile load from f16 row-major X: per-lane K run [hi*16..+15].
__device__ __forceinline__ h16 loadA(const _Float16* __restrict__ row, int hi) {
  return cat16(*(const h8*)(row + hi * 16), *(const h8*)(row + hi * 16 + 8));
}

// ---------------------------------------------------------------------------
// Kernel 0: elementwise fp32 -> f16 (RNE), 8 elems/thread, b128 in/out.
// ---------------------------------------------------------------------------
__global__ __launch_bounds__(256) void cvt_f16_kernel(
    const float* __restrict__ src, _Float16* __restrict__ dst, int n) {
  const int i = (blockIdx.x * 256 + threadIdx.x) * 8;
  if (i + 8 <= n) {
    float4 a = *(const float4*)(src + i);
    float4 b = *(const float4*)(src + i + 4);
    h8 r;
    r[0] = (_Float16)a.x; r[1] = (_Float16)a.y; r[2] = (_Float16)a.z; r[3] = (_Float16)a.w;
    r[4] = (_Float16)b.x; r[5] = (_Float16)b.y; r[6] = (_Float16)b.z; r[7] = (_Float16)b.w;
    *(h8*)(dst + i) = r;
  }
}

// ---------------------------------------------------------------------------
// Kernel 1: Y = X @ W^T + b  (X f16 [8192,512], W f16 [512,512]), f16 out.
// vmode==0: dst[b][h][s][d]  (Qh / Kh layout)
// vmode==1: dst[b][h][d][s]  (V transposed, so P*V B-tiles are contiguous)
// Each wave: 32 rows x 64 cols (B tiles reused across 2 A tiles);
// block = 4 waves (128 rows).
// ---------------------------------------------------------------------------
__global__ __launch_bounds__(128) void qkv_proj_kernel(
    const _Float16* __restrict__ X, const _Float16* __restrict__ W,
    const float* __restrict__ bias, _Float16* __restrict__ dst, int vmode) {
  const int lane = threadIdx.x & 31;
  const int wave = threadIdx.x >> 5;
  const int lo = lane & 15;
  const int hi = lane >> 4;
  const int jBase = blockIdx.x * 64;
  const int mBase = (blockIdx.y * 4 + wave) * 32;

  f8v acc[2][4];
#pragma unroll
  for (int mt = 0; mt < 2; ++mt)
#pragma unroll
    for (int t = 0; t < 4; ++t) acc[mt][t] = f8zero();

  const _Float16* x0 = X + (size_t)(mBase + lo) * D_MODELC;
  const _Float16* x1 = X + (size_t)(mBase + 16 + lo) * D_MODELC;

  for (int kk = 0; kk < D_MODELC; kk += 32) {
    h16 a0 = loadA(x0 + kk, hi);
    h16 a1 = loadA(x1 + kk, hi);
#pragma unroll
    for (int t = 0; t < 4; ++t) {
      // b[i] = W^T[k][n] = W[jBase+t*16+lo][kk + hi*16 + i] (32B contiguous)
      h16 b = *(const h16*)(W + (size_t)(jBase + t * 16 + lo) * D_MODELC + kk +
                            hi * 16);
      acc[0][t] = wmma16(a0, b, acc[0][t]);
      acc[1][t] = wmma16(a1, b, acc[1][t]);
    }
  }

#pragma unroll
  for (int t = 0; t < 4; ++t) {
    const int j = jBase + t * 16 + lo;  // output feature
    const float bv = bias[j];
    const int h = j >> 6;  // head
    const int d = j & 63;  // dim within head
#pragma unroll
    for (int mt = 0; mt < 2; ++mt) {
#pragma unroll
      for (int r = 0; r < 8; ++r) {
        const int mrow = mBase + mt * 16 + hi * 8 + r;  // C/D row = hi*8 + r
        const int bb = mrow >> 12;                      // / 4096
        const int s = mrow & (SEQC - 1);
        const float val = acc[mt][t][r] + bv;
        size_t addr;
        if (vmode)
          addr = (((size_t)(bb * HEADSC + h)) * DKC + d) * SEQC + s;  // Vt
        else
          addr = (((size_t)(bb * HEADSC + h)) * SEQC + s) * DKC + d;  // Qh/Kh
        dst[addr] = (_Float16)val;
      }
    }
  }
}

// ---------------------------------------------------------------------------
// Kernel 2: flash attention.  Block = 4 waves = 64 q-rows for one (b,h).
// Each wave owns 16 q-rows x 64 dims; loops k in blocks of 32.
// grid = (HEADS, SEQ/64, BATCH) so heads sharing mask rows are adjacent (L2).
// ---------------------------------------------------------------------------
__global__ __launch_bounds__(128) void flash_attn_kernel(
    const _Float16* __restrict__ Qh, const _Float16* __restrict__ Kh,
    const _Float16* __restrict__ Vt, const int* __restrict__ mask,
    _Float16* __restrict__ attn_out) {
  __shared__ _Float16 lds[4 * 16 * 32];  // 1KB per wave: P tile staging

  const int lane = threadIdx.x & 31;
  const int wave = threadIdx.x >> 5;
  const int lo = lane & 15;
  const int hi = lane >> 4;
  const int h = blockIdx.x;
  const int b = blockIdx.z;
  const int qBase = blockIdx.y * 64 + wave * 16;

  const _Float16* Qp = Qh + ((size_t)(b * HEADSC + h) * SEQC) * DKC;
  const _Float16* Kp = Kh + ((size_t)(b * HEADSC + h) * SEQC) * DKC;
  const _Float16* Vp = Vt + ((size_t)(b * HEADSC + h) * DKC) * SEQC;
  const int* Mp = mask + (size_t)b * SEQC * SEQC;

  // Q A-tiles (d 0..31 and 32..63), resident for the whole loop.
  const _Float16* qrow = Qp + (size_t)(qBase + lo) * DKC;
  h16 qa0 = loadA(qrow, hi);
  h16 qa1 = loadA(qrow + 32, hi);

  f8v o0 = f8zero(), o1 = f8zero(), o2 = f8zero(), o3 = f8zero();
  float m[8], l[8];
#pragma unroll
  for (int r = 0; r < 8; ++r) { m[r] = -3.0e38f; l[r] = 0.0f; }

  _Float16* pbuf = lds + wave * (16 * 32);
  const float scale = 0.125f;  // 1/sqrt(64)

  for (int kb = 0; kb < SEQC; kb += 32) {
    // ---- scores S[16 x 32] = Q(16x64) * K^T(64x32), f32 accum ----
    const _Float16* k0row = Kp + (size_t)(kb + lo) * DKC;       // kpos = kb+n
    const _Float16* k1row = Kp + (size_t)(kb + 16 + lo) * DKC;  // kpos = kb+16+n
    h16 bK00 = *(const h16*)(k0row + hi * 16);       // d 0..31
    h16 bK01 = *(const h16*)(k0row + 32 + hi * 16);  // d 32..63
    h16 bK10 = *(const h16*)(k1row + hi * 16);
    h16 bK11 = *(const h16*)(k1row + 32 + hi * 16);
    f8v s0 = wmma16(qa0, bK00, f8zero());
    s0 = wmma16(qa1, bK01, s0);
    f8v s1 = wmma16(qa0, bK10, f8zero());
    s1 = wmma16(qa1, bK11, s1);

    // ---- scale, mask, online softmax ----
    float rmax[8];
#pragma unroll
    for (int r = 0; r < 8; ++r) {
      const int q = qBase + hi * 8 + r;
      const int* mrow = Mp + (size_t)q * SEQC + kb;
      float a0 = s0[r] * scale;
      float a1 = s1[r] * scale;
      if (mrow[lo] == 0) a0 = -1.0e9f;
      if (mrow[16 + lo] == 0) a1 = -1.0e9f;
      s0[r] = a0;
      s1[r] = a1;
      float mx = fmaxf(a0, a1);  // xor<16 stays inside the 16-lane row group
      mx = fmaxf(mx, __shfl_xor(mx, 1));
      mx = fmaxf(mx, __shfl_xor(mx, 2));
      mx = fmaxf(mx, __shfl_xor(mx, 4));
      mx = fmaxf(mx, __shfl_xor(mx, 8));
      rmax[r] = mx;
    }
    float alpha[8];
#pragma unroll
    for (int r = 0; r < 8; ++r) {
      const float mn = fmaxf(m[r], rmax[r]);
      alpha[r] = __expf(m[r] - mn);
      m[r] = mn;
    }
#pragma unroll
    for (int r = 0; r < 8; ++r) {
      const float e0 = __expf(s0[r] - m[r]);
      const float e1 = __expf(s1[r] - m[r]);
      float rs = e0 + e1;
      rs += __shfl_xor(rs, 1);
      rs += __shfl_xor(rs, 2);
      rs += __shfl_xor(rs, 4);
      rs += __shfl_xor(rs, 8);
      l[r] = l[r] * alpha[r] + rs;
      // stage P (row-major 16x32 f16) for layout conversion C -> A
      pbuf[(hi * 8 + r) * 32 + lo] = (_Float16)e0;
      pbuf[(hi * 8 + r) * 32 + 16 + lo] = (_Float16)e1;
    }
    // rescale running output accumulators
#pragma unroll
    for (int r = 0; r < 8; ++r) {
      o0[r] *= alpha[r];
      o1[r] *= alpha[r];
      o2[r] *= alpha[r];
      o3[r] *= alpha[r];
    }

    // Same-wave DS ops are in-order; wait + compiler barrier before re-read.
    asm volatile("s_wait_dscnt 0" ::: "memory");

    // P as A-tile (16x32 over k)
    h16 pa = cat16(*(const h8*)(pbuf + lo * 32 + hi * 16),
                   *(const h8*)(pbuf + lo * 32 + hi * 16 + 8));

    // ---- O += P(16x32) * V(32x64); V transposed so B-tiles contiguous ----
    const _Float16* v0 = Vp + (size_t)(0 * 16 + lo) * SEQC + kb + hi * 16;
    const _Float16* v1 = Vp + (size_t)(1 * 16 + lo) * SEQC + kb + hi * 16;
    const _Float16* v2 = Vp + (size_t)(2 * 16 + lo) * SEQC + kb + hi * 16;
    const _Float16* v3 = Vp + (size_t)(3 * 16 + lo) * SEQC + kb + hi * 16;
    o0 = wmma16(pa, *(const h16*)v0, o0);
    o1 = wmma16(pa, *(const h16*)v1, o1);
    o2 = wmma16(pa, *(const h16*)v2, o2);
    o3 = wmma16(pa, *(const h16*)v3, o3);
  }

  // Normalize and store attention output as [b][s][h*64 + d] (f16).
#pragma unroll
  for (int r = 0; r < 8; ++r) {
    const float inv = 1.0f / l[r];
    const int q = qBase + hi * 8 + r;
    _Float16* orow = attn_out + ((size_t)(b * SEQC + q)) * D_MODELC + h * DKC;
    orow[0 * 16 + lo] = (_Float16)(o0[r] * inv);
    orow[1 * 16 + lo] = (_Float16)(o1[r] * inv);
    orow[2 * 16 + lo] = (_Float16)(o2[r] * inv);
    orow[3 * 16 + lo] = (_Float16)(o3[r] * inv);
  }
}

// ---------------------------------------------------------------------------
// Kernel 3: final projection Y = X @ W0^T + b0, X/W0 f16, Y fp32 (d_out).
// Each wave: 32 rows x 64 cols, like kernel 1.
// ---------------------------------------------------------------------------
__global__ __launch_bounds__(128) void out_proj_kernel(
    const _Float16* __restrict__ X, const _Float16* __restrict__ W,
    const float* __restrict__ bias, float* __restrict__ Y) {
  const int lane = threadIdx.x & 31;
  const int wave = threadIdx.x >> 5;
  const int lo = lane & 15;
  const int hi = lane >> 4;
  const int jBase = blockIdx.x * 64;
  const int mBase = (blockIdx.y * 4 + wave) * 32;

  f8v acc[2][4];
#pragma unroll
  for (int mt = 0; mt < 2; ++mt)
#pragma unroll
    for (int t = 0; t < 4; ++t) acc[mt][t] = f8zero();

  const _Float16* x0 = X + (size_t)(mBase + lo) * D_MODELC;
  const _Float16* x1 = X + (size_t)(mBase + 16 + lo) * D_MODELC;

  for (int kk = 0; kk < D_MODELC; kk += 32) {
    h16 a0 = loadA(x0 + kk, hi);
    h16 a1 = loadA(x1 + kk, hi);
#pragma unroll
    for (int t = 0; t < 4; ++t) {
      h16 b = *(const h16*)(W + (size_t)(jBase + t * 16 + lo) * D_MODELC + kk +
                            hi * 16);
      acc[0][t] = wmma16(a0, b, acc[0][t]);
      acc[1][t] = wmma16(a1, b, acc[1][t]);
    }
  }

#pragma unroll
  for (int t = 0; t < 4; ++t) {
    const int j = jBase + t * 16 + lo;
    const float bv = bias[j];
#pragma unroll
    for (int mt = 0; mt < 2; ++mt) {
#pragma unroll
      for (int r = 0; r < 8; ++r) {
        const int mrow = mBase + mt * 16 + hi * 8 + r;
        Y[(size_t)mrow * D_MODELC + j] = acc[mt][t][r] + bv;
      }
    }
  }
}

// ---------------------------------------------------------------------------
extern "C" void kernel_launch(void* const* d_in, const int* in_sizes, int n_in,
                              void* d_out, int out_size, void* d_ws,
                              size_t ws_size, hipStream_t stream) {
  const float* q = (const float*)d_in[0];
  const float* k = (const float*)d_in[1];
  const float* v = (const float*)d_in[2];
  const float* wq = (const float*)d_in[3];
  const float* bq = (const float*)d_in[4];
  const float* wk = (const float*)d_in[5];
  const float* bk = (const float*)d_in[6];
  const float* wv = (const float*)d_in[7];
  const float* bv = (const float*)d_in[8];
  const float* w0 = (const float*)d_in[9];
  const float* b0 = (const float*)d_in[10];
  const int* mask = (const int*)d_in[11];
  float* out = (float*)d_out;

  const size_t actElems = (size_t)BATCHC * SEQC * D_MODELC;       // 4.19M
  const size_t headElems = (size_t)BATCHC * HEADSC * SEQC * DKC;  // 4.19M
  const size_t wElems = (size_t)D_MODELC * D_MODELC;              // 262144

  _Float16* Qh = (_Float16*)d_ws;
  _Float16* Kh = Qh + headElems;
  _Float16* Vt = Kh + headElems;
  _Float16* At = Vt + headElems;  // attention output [B,S,512] f16
  _Float16* Xq = At + actElems;   // f16 copies of activations
  _Float16* Xk = Xq + actElems;
  _Float16* Xv = Xk + actElems;
  _Float16* Wq = Xv + actElems;  // f16 copies of weights
  _Float16* Wk = Wq + wElems;
  _Float16* Wv = Wk + wElems;
  _Float16* W0 = Wv + wElems;

  // ---- pre-convert operands to f16 once (activations reused 8x, weights 64x)
  {
    dim3 cb(256);
    dim3 cga((unsigned)((actElems / 8 + 255) / 256));
    dim3 cgw((unsigned)((wElems / 8 + 255) / 256));
    cvt_f16_kernel<<<cga, cb, 0, stream>>>(q, Xq, (int)actElems);
    cvt_f16_kernel<<<cga, cb, 0, stream>>>(k, Xk, (int)actElems);
    cvt_f16_kernel<<<cga, cb, 0, stream>>>(v, Xv, (int)actElems);
    cvt_f16_kernel<<<cgw, cb, 0, stream>>>(wq, Wq, (int)wElems);
    cvt_f16_kernel<<<cgw, cb, 0, stream>>>(wk, Wk, (int)wElems);
    cvt_f16_kernel<<<cgw, cb, 0, stream>>>(wv, Wv, (int)wElems);
    cvt_f16_kernel<<<cgw, cb, 0, stream>>>(w0, W0, (int)wElems);
  }

  dim3 blk(128);
  dim3 g1(D_MODELC / 64, (BATCHC * SEQC) / 128);  // (8, 64)
  qkv_proj_kernel<<<g1, blk, 0, stream>>>(Xq, Wq, bq, Qh, 0);
  qkv_proj_kernel<<<g1, blk, 0, stream>>>(Xk, Wk, bk, Kh, 0);
  qkv_proj_kernel<<<g1, blk, 0, stream>>>(Xv, Wv, bv, Vt, 1);

  dim3 g2(HEADSC, SEQC / 64, BATCHC);  // heads fastest: share mask rows in L2
  flash_attn_kernel<<<g2, blk, 0, stream>>>(Qh, Kh, Vt, mask, At);

  out_proj_kernel<<<g1, blk, 0, stream>>>(At, W0, b0, out);
  (void)in_sizes; (void)n_in; (void)out_size; (void)ws_size;
}
